// CAttention_67199058313545
// MI455X (gfx1250) — matrix operand order repaired and checked by reading, not compiled, not run on hardware
//
#include <hip/hip_runtime.h>

#define HEADS 8
#define DIMH  32
#define CCH   256
#define HH    56
#define WWD   56
#define PP    (HH * WWD)      // 3136
#define NBAT  2
#define NPR   (NBAT * PP)     // 6272 rows total
#define C3    (3 * CCH)       // 768

typedef __bf16 bf16_t;
typedef bf16_t v16bf __attribute__((ext_vector_type(16)));
typedef float  v8f   __attribute__((ext_vector_type(8)));

union TileBuf {
  v16bf v;
  uint4 q[2];
};

__device__ __forceinline__ unsigned short f2bf(float f) {
  union { float f; unsigned u; } v; v.f = f;
  unsigned r = v.u + 0x7FFFu + ((v.u >> 16) & 1u);
  return (unsigned short)(r >> 16);
}

// ---------------------------------------------------------------------------
// Stage 0a: transpose (N,C,P) fp32 -> (N*P, C) bf16 row-major panels
// ---------------------------------------------------------------------------
__global__ __launch_bounds__(256) void prep_inputs(const float* __restrict__ q,
                                                   const float* __restrict__ k,
                                                   unsigned short* __restrict__ Xq,
                                                   unsigned short* __restrict__ Xk) {
  int t = blockIdx.x * 256 + threadIdx.x;        // over N*C*P
  if (t >= NBAT * CCH * PP) return;
  int p = t % PP;
  int c = (t / PP) % CCH;
  int n = t / (CCH * PP);
  size_t dst = (size_t)(n * PP + p) * CCH + c;
  Xq[dst] = f2bf(q[t]);
  Xk[dst] = f2bf(k[t]);
}

// ---------------------------------------------------------------------------
// Stage 0b: bf16 weights + fused biases.
//   bias[r] = b[r] + (r>=256 ? pos * rowsum(W[r,:]) : 0)
// (pos bias folds into a constant per output channel; OOB neighbors equal it)
// ---------------------------------------------------------------------------
__global__ __launch_bounds__(256) void prep_weights(const float* __restrict__ inw,
                                                    const float* __restrict__ inb,
                                                    const float* __restrict__ pos,
                                                    const float* __restrict__ outw,
                                                    unsigned short* __restrict__ Wbf,
                                                    unsigned short* __restrict__ OWbf,
                                                    float* __restrict__ bias) {
  int r = blockIdx.x * 256 + threadIdx.x;
  if (r < C3) {
    float s = 0.f;
    for (int c = 0; c < CCH; ++c) {
      float w = inw[(size_t)r * CCH + c];
      s += w;
      Wbf[(size_t)r * CCH + c] = f2bf(w);
    }
    bias[r] = inb[r] + ((r >= CCH) ? pos[0] * s : 0.f);
  } else if (r < C3 + CCH) {
    int ro = r - C3;
    for (int c = 0; c < CCH; ++c)
      OWbf[(size_t)ro * CCH + c] = f2bf(outw[(size_t)ro * CCH + c]);
  }
}

// ---------------------------------------------------------------------------
// WMMA GEMM: D[row, col] = sum_k A[row,k] * Bw[col,k] + bias[col]
// One wave computes a 16x64 tile: 4 accumulators share one A fetch per k-step
// (4x A-reuse -> 4x fewer A-side L2 reads, 4 v_wmma per loop iteration).
// mode 0: write D row-major stride 768 (QKV); A = (colBase<256 ? Aq : Ak)
//         (the 256-col Q/K boundary is 64-aligned, so per-group select is exact)
// mode 1: write D transposed into NCHW fp32 output (out-projection)
// ---------------------------------------------------------------------------
__global__ __launch_bounds__(128) void gemm_wmma(const unsigned short* __restrict__ Aq,
                                                 const unsigned short* __restrict__ Ak,
                                                 const unsigned short* __restrict__ Bw,
                                                 const float* __restrict__ bias,
                                                 float* __restrict__ Dout,
                                                 int colGroups, int mode) {
  int lane = threadIdx.x & 31;
  int wave = threadIdx.x >> 5;
  int tile = blockIdx.x * 4 + wave;
  int rowTile = (tile / colGroups) * 16;
  int colBase = (tile % colGroups) * 64;
  if (rowTile >= NPR) return;

  bool hi  = lane >= 16;
  int  l16 = lane & 15;

  const unsigned short* A = (mode == 0 && colBase >= CCH) ? Ak : Aq;
  const unsigned short* arow = A + (size_t)(rowTile + l16) * CCH;
  const unsigned short* brow[4];
#pragma unroll
  for (int j = 0; j < 4; ++j)
    brow[j] = Bw + (size_t)(colBase + j * 16 + l16) * CCH;

  v8f acc0 = {}, acc1 = {}, acc2 = {}, acc3 = {};
#pragma unroll
  for (int kb = 0; kb < CCH; kb += 32) {
    TileBuf a, b0, b1, b2, b3;
    // A (16x32 bf16): lanes 0-15 hold K {kb..kb+15}; lanes 16-31 hold K {kb+16..kb+31}
    int ka = kb + (hi ? 16 : 0);
    a.q[0] = *(const uint4*)(arow + ka);
    a.q[1] = *(const uint4*)(arow + ka + 8);
    // B (32x16 bf16): lanes 0-15 hold K {kb..kb+15}; lanes 16-31 hold K {kb+16..kb+31}
    int kbb = kb + (hi ? 16 : 0);
    b0.q[0] = *(const uint4*)(brow[0] + kbb);
    b0.q[1] = *(const uint4*)(brow[0] + kbb + 8);
    b1.q[0] = *(const uint4*)(brow[1] + kbb);
    b1.q[1] = *(const uint4*)(brow[1] + kbb + 8);
    b2.q[0] = *(const uint4*)(brow[2] + kbb);
    b2.q[1] = *(const uint4*)(brow[2] + kbb + 8);
    b3.q[0] = *(const uint4*)(brow[3] + kbb);
    b3.q[1] = *(const uint4*)(brow[3] + kbb + 8);
    if (kb + 32 < CCH)
      __builtin_prefetch(arow + kb + 32, 0, 1);
    acc0 = __builtin_amdgcn_wmma_f32_16x16x32_bf16(false, a.v, false, b0.v,
                                                   (short)0, acc0, false, false);
    acc1 = __builtin_amdgcn_wmma_f32_16x16x32_bf16(false, a.v, false, b1.v,
                                                   (short)0, acc1, false, false);
    acc2 = __builtin_amdgcn_wmma_f32_16x16x32_bf16(false, a.v, false, b2.v,
                                                   (short)0, acc2, false, false);
    acc3 = __builtin_amdgcn_wmma_f32_16x16x32_bf16(false, a.v, false, b3.v,
                                                   (short)0, acc3, false, false);
  }

  // D layout: VGPR r -> (M = r + (hi?8:0), N = l16)
  v8f accs[4] = {acc0, acc1, acc2, acc3};
#pragma unroll
  for (int j = 0; j < 4; ++j) {
#pragma unroll
    for (int r = 0; r < 8; ++r) {
      int row = rowTile + r + (hi ? 8 : 0);
      int col = colBase + j * 16 + l16;
      float v = accs[j][r] + bias[col];
      if (mode == 0) {
        Dout[(size_t)row * C3 + col] = v;
      } else {
        int n  = row / PP;
        int pl = row % PP;
        Dout[(size_t)n * CCH * PP + (size_t)col * PP + pl] = v;
      }
    }
  }
}

// ---------------------------------------------------------------------------
// Stage 2: local attention, one thread per (pixel, head), flash-style
// online softmax over the 7x7 neighborhood. OOB neighbors use the constant
// bias vectors (exactly what the zero-padded + pos-bias reference produces).
// ---------------------------------------------------------------------------
__global__ __launch_bounds__(256) void attn_fused(const float* __restrict__ QKV,
                                                  const float* __restrict__ bias,
                                                  unsigned short* __restrict__ Obf) {
  int t = blockIdx.x * 256 + threadIdx.x;        // over NPR * HEADS
  if (t >= NPR * HEADS) return;
  int h   = t & (HEADS - 1);
  int row = t >> 3;
  int n  = row / PP;
  int pl = row % PP;
  int y  = pl / WWD;
  int x  = pl % WWD;

  const float* qp = QKV + (size_t)row * C3 + h * DIMH;
  float qv[DIMH];
#pragma unroll
  for (int i = 0; i < DIMH; ++i) qv[i] = qp[i];

  const float* bk = bias + CCH + h * DIMH;
  const float* bv = bias + 2 * CCH + h * DIMH;

  float m = -3.4e38f, l = 0.f;
  float acc[DIMH];
#pragma unroll
  for (int i = 0; i < DIMH; ++i) acc[i] = 0.f;
  const float scale = 0.17677669529663687f;      // 1/sqrt(32)

  for (int dy = -3; dy <= 3; ++dy) {
    for (int dx = -3; dx <= 3; ++dx) {
      int yy = y + dy, xx = x + dx;
      const float *kp, *vp;
      if (yy >= 0 && yy < HH && xx >= 0 && xx < WWD) {
        size_t r2 = (size_t)(n * PP + yy * WWD + xx) * C3;
        kp = QKV + r2 + CCH + h * DIMH;
        vp = QKV + r2 + 2 * CCH + h * DIMH;
      } else {
        kp = bk; vp = bv;
      }
      float s = 0.f;
#pragma unroll
      for (int i = 0; i < DIMH; ++i) s += qv[i] * kp[i];
      s *= scale;
      float mn = fmaxf(m, s);
      float cf = __expf(m - mn);
      float w  = __expf(s - mn);
      l = l * cf + w;
#pragma unroll
      for (int i = 0; i < DIMH; ++i) acc[i] = acc[i] * cf + w * vp[i];
      m = mn;
    }
  }

  float inv = 1.f / l;
  unsigned short* op = Obf + (size_t)row * CCH + h * DIMH;
#pragma unroll
  for (int i = 0; i < DIMH; ++i) op[i] = f2bf(acc[i] * inv);
}

// ---------------------------------------------------------------------------
extern "C" void kernel_launch(void* const* d_in, const int* in_sizes, int n_in,
                              void* d_out, int out_size, void* d_ws, size_t ws_size,
                              hipStream_t stream) {
  const float* queries = (const float*)d_in[0];
  const float* key     = (const float*)d_in[1];
  const float* pos     = (const float*)d_in[2];
  const float* inw     = (const float*)d_in[3];
  const float* inb     = (const float*)d_in[4];
  const float* outw    = (const float*)d_in[5];
  const float* outb    = (const float*)d_in[6];
  float* out = (float*)d_out;

  // workspace layout (element offsets)
  unsigned short* Xq   = (unsigned short*)d_ws;               // NPR*256 bf16
  unsigned short* Xk   = Xq   + (size_t)NPR * CCH;            // NPR*256 bf16
  unsigned short* Wbf  = Xk   + (size_t)NPR * CCH;            // 768*256 bf16
  unsigned short* OWbf = Wbf  + (size_t)C3 * CCH;             // 256*256 bf16
  float*          bias = (float*)(OWbf + (size_t)CCH * CCH);  // 768 f32
  float*          QKV  = bias + C3;                           // NPR*768 f32
  unsigned short* Obf  = (unsigned short*)(QKV + (size_t)NPR * C3); // NPR*256 bf16

  // Stage 0: transpose inputs to bf16 row panels; bf16 weights + fused biases
  prep_inputs<<<(NBAT * CCH * PP + 255) / 256, 256, 0, stream>>>(queries, key, Xq, Xk);
  prep_weights<<<4, 256, 0, stream>>>(inw, inb, pos, outw, Wbf, OWbf, bias);

  // Stage 1: [Q|K|V] projection GEMM, (6272 x 768): 392 row-tiles x 12 col-groups
  gemm_wmma<<<(NPR / 16) * (C3 / 64) / 4, 128, 0, stream>>>(
      Xq, Xk, Wbf, bias, QKV, C3 / 64, 0);

  // Stage 2: fused local attention (online softmax over 49 neighbors)
  attn_fused<<<NPR * HEADS / 256, 256, 0, stream>>>(QKV, bias, Obf);

  // Stage 3: out-projection GEMM, (6272 x 256): 392 row-tiles x 4 col-groups,
  // writes transposed NCHW fp32 directly
  gemm_wmma<<<(NPR / 16) * (CCH / 64) / 4, 128, 0, stream>>>(
      Obf, Obf, OWbf, outb, out, CCH / 64, 1);
}
